// SpanPrunerHOI_3444563771560
// MI455X (gfx1250) — matrix operand order repaired and checked by reading, not compiled, not run on hardware
//
#include <hip/hip_runtime.h>
#include <hip/hip_bf16.h>
#include <math.h>

// ---------------- problem constants (match reference module constants) -------
#define MCAND   30000
#define KDIM    3092
#define NDIM    1000
#define UNARYF  20
#define MAXW    30
#define NW      1024
#define PADW    31            // MAX_W + 1
#define NTOP    409           // int(min(3900, 1024*0.4))
#define SORT_N  32768

// GEMM blocking
#define MBLK    128
#define NBLK    128
#define KSTEP   32
#define KTILES  ((KDIM + KSTEP - 1) / KSTEP)   // 97
#define LDS_STRIDE 40          // 32 + 8 pad bf16 elems -> 80B rows (16B aligned)
#define TILE_ELEMS (MBLK * LDS_STRIDE)

typedef __attribute__((ext_vector_type(16))) __bf16  bf16x16;
typedef __attribute__((ext_vector_type(8)))  float   f32x8;
typedef __attribute__((ext_vector_type(4)))  unsigned int u32x4;

union Frag16 { bf16x16 v; u32x4 q[2]; };

__device__ __forceinline__ unsigned short f2bf(float f) {
    union { float f; unsigned int u; } a; a.f = f;
    unsigned int r = (a.u + 0x7FFFu + ((a.u >> 16) & 1u)) >> 16;   // RNE
    return (unsigned short)r;
}

__device__ __forceinline__ unsigned long long pack4(float x, float y, float z, float w) {
    return (unsigned long long)f2bf(x)
         | ((unsigned long long)f2bf(y) << 16)
         | ((unsigned long long)f2bf(z) << 32)
         | ((unsigned long long)f2bf(w) << 48);
}

// ---------------------------------------------------------------------------
// Cooperative staging of one K-tile into LDS (bf16; B stored transposed [n][k])
// fast==true: interior tile, branch-free vectorized path.
// ---------------------------------------------------------------------------
__device__ __forceinline__ void stage_tile(const float* __restrict__ A,
                                           const float* __restrict__ W1,
                                           int m0, int n0, int k0, int tid,
                                           unsigned short* __restrict__ As,
                                           unsigned short* __restrict__ Bs,
                                           bool fast) {
    if (fast) {
        // ---- A: thread -> (row = tid>>1, 16-col segment), 4x float4 loads ----
        const int row = tid >> 1;
        const int cs  = (tid & 1) * 16;
        const float* ap = A + (size_t)(m0 + row) * KDIM + k0 + cs;
        unsigned short* as = As + row * LDS_STRIDE + cs;
        #pragma unroll
        for (int q = 0; q < 4; ++q) {
            float4 v = *(const float4*)(ap + q * 4);
            *(unsigned long long*)(as + q * 4) = pack4(v.x, v.y, v.z, v.w);
        }
        // ---- B: thread -> (4 k-rows, 4-col n segment); 4x4 register transpose ----
        const int kb = (tid >> 5) * 4;        // k base: 0..28
        const int ns = (tid & 31) * 4;        // n base: 0..124
        float rr[4][4];
        #pragma unroll
        for (int i = 0; i < 4; ++i) {
            float4 v = *(const float4*)(W1 + (size_t)(k0 + kb + i) * NDIM + n0 + ns);
            rr[i][0] = v.x; rr[i][1] = v.y; rr[i][2] = v.z; rr[i][3] = v.w;
        }
        #pragma unroll
        for (int j = 0; j < 4; ++j) {
            *(unsigned long long*)(Bs + (ns + j) * LDS_STRIDE + kb) =
                pack4(rr[0][j], rr[1][j], rr[2][j], rr[3][j]);
        }
    } else {
        // ---- guarded scalar path (edge blocks / last K-tile only) ----
        const int a_col  = tid & 31;
        const int a_row0 = (tid >> 5) * 16;
        #pragma unroll 4
        for (int i = 0; i < 16; ++i) {
            int row = a_row0 + i;
            int gm = m0 + row, gk = k0 + a_col;
            float v = (gm < MCAND && gk < KDIM) ? A[(size_t)gm * KDIM + gk] : 0.f;
            As[row * LDS_STRIDE + a_col] = f2bf(v);
        }
        const int b_n  = tid & 127;
        const int b_k0 = (tid >> 7) * 16;
        #pragma unroll 4
        for (int i = 0; i < 16; ++i) {
            int k = b_k0 + i;
            int gk = k0 + k, gn = n0 + b_n;
            float v = (gk < KDIM && gn < NDIM) ? W1[(size_t)gk * NDIM + gn] : 0.f;
            Bs[b_n * LDS_STRIDE + k] = f2bf(v);
        }
    }
}

// ---------------------------------------------------------------------------
// Kernel 1: width prior FFNN  (tiny: [30,20]@[20,1000] -> relu -> dot ww2)
// ---------------------------------------------------------------------------
__global__ void k_width(const float* __restrict__ wemb, const float* __restrict__ Ww1,
                        const float* __restrict__ bw1, const float* __restrict__ ww2,
                        const float* __restrict__ bw2, float* __restrict__ wscore) {
    __shared__ float red[256];
    int w = blockIdx.x;
    float part = 0.f;
    for (int u = threadIdx.x; u < NDIM; u += 256) {
        float acc = bw1[u];
        #pragma unroll
        for (int k = 0; k < UNARYF; ++k)
            acc += wemb[w * UNARYF + k] * Ww1[k * NDIM + u];
        part += fmaxf(acc, 0.f) * ww2[u];
    }
    red[threadIdx.x] = part;
    __syncthreads();
    for (int s = 128; s > 0; s >>= 1) {
        if (threadIdx.x < s) red[threadIdx.x] += red[threadIdx.x + s];
        __syncthreads();
    }
    if (threadIdx.x == 0) wscore[w] = red[0] + bw2[0];
}

// ---------------------------------------------------------------------------
// Kernel 2: scores[i] = b2 + width_score[width_idx[i]]  (GEMM atomically adds)
// ---------------------------------------------------------------------------
__global__ void k_init_scores(const float* __restrict__ wscore, const float* __restrict__ b2,
                              const int* __restrict__ widx, float* __restrict__ scores) {
    int i = blockIdx.x * blockDim.x + threadIdx.x;
    if (i < MCAND) scores[i] = b2[0] + wscore[widx[i]];
}

// ---------------------------------------------------------------------------
// Kernel 3: fused span-score GEMM:
//   scores[m] += sum_n w2[n] * relu( sum_k A[m,k]*W1[k,n] + b1[n] )
// bf16 WMMA 16x16x32, fp32 accumulate, double-buffered LDS, 1 barrier/K-step.
// ---------------------------------------------------------------------------
__global__ void __launch_bounds__(256)
k_span_gemm(const float* __restrict__ A, const float* __restrict__ W1,
            const float* __restrict__ b1, const float* __restrict__ w2,
            float* __restrict__ scores) {
    __shared__ unsigned short As[2][TILE_ELEMS];   // [buf][row(m)][k] bf16
    __shared__ unsigned short Bs[2][TILE_ELEMS];   // [buf][col(n)][k] bf16 (transposed)

    const int tid  = threadIdx.x;
    const int lane = tid & 31;
    const int wave = tid >> 5;
    const int mw   = wave >> 1;        // 0..3 -> 2 M-tiles each
    const int nw   = wave & 1;         // 0..1 -> 4 N-tiles each
    const int half = lane >> 4;        // 0: lanes 0-15, 1: lanes 16-31
    const int nloc = lane & 15;

    const int m0 = blockIdx.x * MBLK;
    const int n0 = blockIdx.y * NBLK;
    const bool mn_interior = (m0 + MBLK <= MCAND) && (n0 + NBLK <= NDIM);

    f32x8 acc[2][4] = {};

    stage_tile(A, W1, m0, n0, 0, tid, As[0], Bs[0],
               mn_interior && (KSTEP <= KDIM));
    __syncthreads();

    for (int kt = 0; kt < KTILES; ++kt) {
        const int cur = kt & 1;

        // prefetch tile kt+2 into cache (global_prefetch_b8)
        if (kt + 2 < KTILES) {
            const int kp = (kt + 2) * KSTEP;
            __builtin_prefetch(A  + (size_t)(m0 + (tid >> 1)) * KDIM + kp + (tid & 1) * 16, 0, 1);
            __builtin_prefetch(W1 + (size_t)(kp + (tid >> 5) * 4) * NDIM + n0 + (tid & 31) * 4, 0, 1);
        }
        // stage next tile into the other buffer (overlaps with WMMA below)
        if (kt + 1 < KTILES) {
            const int kn = (kt + 1) * KSTEP;
            stage_tile(A, W1, m0, n0, kn, tid, As[cur ^ 1], Bs[cur ^ 1],
                       mn_interior && (kn + KSTEP <= KDIM));
        }

        // ---- build fragments (ISA 16-bit A 16x32 / B 32x16 layouts) ----
        Frag16 af[2], bf[4];
        #pragma unroll
        for (int mt = 0; mt < 2; ++mt) {
            const unsigned short* p = &As[cur][(mw * 32 + mt * 16 + nloc) * LDS_STRIDE];
            int c0 = half * 8;                 // lanes>=16 start at K=8
            af[mt].q[0] = *(const u32x4*)(p + c0);        // K = c0..c0+7
            af[mt].q[1] = *(const u32x4*)(p + c0 + 16);   // K = c0+16..c0+23
        }
        #pragma unroll
        for (int nt = 0; nt < 4; ++nt) {
            const unsigned short* p =
                &Bs[cur][(nw * 64 + nt * 16 + nloc) * LDS_STRIDE + half * 8]; // K = h*8.., h*8+16..
            bf[nt].q[0] = *(const u32x4*)p;
            bf[nt].q[1] = *(const u32x4*)(p + 16);
        }

        // ---- 8 WMMAs per wave per K-step ----
        #pragma unroll
        for (int mt = 0; mt < 2; ++mt)
            #pragma unroll
            for (int nt = 0; nt < 4; ++nt)
                acc[mt][nt] = __builtin_amdgcn_wmma_f32_16x16x32_bf16(
                    false, af[mt].v, false, bf[nt].v,
                    (short)0, acc[mt][nt], false, false);

        __syncthreads();   // next-buffer writes done; current buffer free to reuse
    }

    // ---- epilogue: relu(acc + b1) * w2, reduce over N, atomicAdd scores ----
    float bb[4], wv[4];
    #pragma unroll
    for (int nt = 0; nt < 4; ++nt) {
        int N = n0 + nw * 64 + nt * 16 + nloc;
        bool ok = (N < NDIM);
        bb[nt] = ok ? b1[N] : 0.f;
        wv[nt] = ok ? w2[N] : 0.f;
    }
    #pragma unroll
    for (int mt = 0; mt < 2; ++mt) {
        #pragma unroll
        for (int r = 0; r < 8; ++r) {
            float s = 0.f;
            #pragma unroll
            for (int nt = 0; nt < 4; ++nt) {
                float h = acc[mt][nt][r] + bb[nt];
                s += fmaxf(h, 0.f) * wv[nt];
            }
            // reduce across the 16 lanes of this half (C rows live per-half)
            #pragma unroll
            for (int off = 1; off < 16; off <<= 1) s += __shfl_xor(s, off, 32);
            int Mrow = m0 + mw * 32 + mt * 16 + r + half * 8;
            if (nloc == 0 && Mrow < MCAND) atomicAdd(&scores[Mrow], s);
        }
    }
}

// ---------------------------------------------------------------------------
// Kernel 4/5: bitonic argsort of -scores (ascending -score == descending score)
// ---------------------------------------------------------------------------
__global__ void k_sort_init(const float* __restrict__ scores,
                            float* __restrict__ keys, int* __restrict__ vals) {
    int i = blockIdx.x * blockDim.x + threadIdx.x;
    if (i < SORT_N) {
        if (i < MCAND) { keys[i] = -scores[i]; vals[i] = i; }
        else           { keys[i] = INFINITY;   vals[i] = 0x7FFFFFFF; }
    }
}

__global__ void k_bitonic(float* __restrict__ keys, int* __restrict__ vals,
                          unsigned j, unsigned k) {
    unsigned i = blockIdx.x * blockDim.x + threadIdx.x;
    unsigned ixj = i ^ j;
    if (ixj > i && i < SORT_N) {
        bool up = ((i & k) == 0);
        float a = keys[i], b = keys[ixj];
        if (up ? (a > b) : (a < b)) {
            keys[i] = b; keys[ixj] = a;
            int va = vals[i]; vals[i] = vals[ixj]; vals[ixj] = va;
        }
    }
}

// ---------------------------------------------------------------------------
// Kernel 6: greedy non-crossing span extraction. One wave32; 31 lanes do the
// crossing test in parallel; early-exit once NTOP spans are taken.
// ---------------------------------------------------------------------------
__global__ void k_greedy(const int* __restrict__ order, const int* __restrict__ starts,
                         const int* __restrict__ ends, int* __restrict__ sel_raw) {
    __shared__ int s2m[NW + PADW];
    __shared__ int e2m[NW + PADW];
    __shared__ int sel[NTOP];
    int t = threadIdx.x;
    for (int i = t; i < NW + PADW; i += 32) { s2m[i] = -1; e2m[i] = -1; }
    __syncthreads();

    int count = 0;
    for (int it = 0; it < MCAND && count < NTOP; ++it) {
        int cand = order[it];
        int s = starts[cand];
        int e = ends[cand];
        bool c = false;
        if (t < PADW) {
            int tok = s + t;
            int me = s2m[tok];
            int ms = e2m[tok];
            c = ((tok > s) & (tok <= e) & (me > e)) |
                ((tok < e) & (ms >= 0) & (ms < s));
        }
        bool cross = __any(c);
        if (!cross) {
            if (t == 0) {
                s2m[s] = max(s2m[s], e);
                e2m[e] = (e2m[e] < 0) ? s : min(e2m[e], s);
                sel[count] = cand;
            }
            ++count;
        }
    }
    __syncthreads();
    if (t == 0) { int f = sel[0]; for (int c2 = count; c2 < NTOP; ++c2) sel[c2] = f; }
    __syncthreads();
    for (int i = t; i < NTOP; i += 32) sel_raw[i] = sel[i];
}

// ---------------------------------------------------------------------------
// Kernel 7: stable sort of the 409 selected spans by key=start*(NW+1)+end
// via O(N^2) rank counting (stable tie-break on original position).
// ---------------------------------------------------------------------------
__global__ void k_ranksort(const int* __restrict__ sel_raw, const int* __restrict__ starts,
                           const int* __restrict__ ends, int* __restrict__ sel_sorted) {
    __shared__ int ks[NTOP];
    __shared__ int ss[NTOP];
    int t = threadIdx.x;
    if (t < NTOP) {
        int s = sel_raw[t];
        ss[t] = s;
        ks[t] = starts[s] * (NW + 1) + ends[s];
    }
    __syncthreads();
    if (t < NTOP) {
        int key = ks[t], r = 0;
        for (int j = 0; j < NTOP; ++j) {
            int kj = ks[j];
            r += (kj < key) || (kj == key && j < t);
        }
        sel_sorted[r] = ss[t];
    }
}

// ---------------------------------------------------------------------------
// Kernel 8: gather all six outputs into d_out (concatenated, 4B elems)
// ---------------------------------------------------------------------------
__global__ void k_gather(const int* __restrict__ sel_sorted, const float* __restrict__ emb,
                         const float* __restrict__ scores, const int* __restrict__ starts,
                         const int* __restrict__ ends, const int* __restrict__ spk,
                         float* __restrict__ out) {
    int b = blockIdx.x;
    int idx = sel_sorted[b];
    int* iout = (int*)out;
    if (threadIdx.x == 0) {
        iout[b]            = idx;                    // sel
        iout[NTOP + b]     = starts[idx];            // pruned_starts
        iout[2 * NTOP + b] = ends[idx];              // pruned_ends
        out[3 * NTOP + (size_t)NTOP * KDIM + b] = scores[idx];            // pruned_scores
        iout[4 * NTOP + NTOP * KDIM + b]        = spk[starts[idx]];       // pruned_speakers
    }
    const float* src = emb + (size_t)idx * KDIM;
    float*       dst = out + 3 * NTOP + (size_t)b * KDIM;                 // pruned_emb
    for (int j = threadIdx.x; j < KDIM; j += blockDim.x) dst[j] = src[j];
}

// ---------------------------------------------------------------------------
extern "C" void kernel_launch(void* const* d_in, const int* in_sizes, int n_in,
                              void* d_out, int out_size, void* d_ws, size_t ws_size,
                              hipStream_t stream) {
    const float* emb    = (const float*)d_in[0];
    const float* W1     = (const float*)d_in[1];
    const float* b1     = (const float*)d_in[2];
    const float* w2     = (const float*)d_in[3];
    const float* b2     = (const float*)d_in[4];
    const float* Ww1    = (const float*)d_in[5];
    const float* bw1    = (const float*)d_in[6];
    const float* ww2    = (const float*)d_in[7];
    const float* bw2    = (const float*)d_in[8];
    const float* wemb   = (const float*)d_in[9];
    const int*   widx   = (const int*)d_in[10];
    const int*   starts = (const int*)d_in[11];
    const int*   ends   = (const int*)d_in[12];
    const int*   spk    = (const int*)d_in[13];
    (void)in_sizes; (void)n_in; (void)out_size; (void)ws_size;

    float* ws         = (float*)d_ws;
    float* wscore     = ws;                       // 32
    float* scores     = ws + 32;                  // 30000 (-> 30032)
    float* keys       = ws + 30064;               // 32768 (-> 62832)
    int*   vals       = (int*)(ws + 62832);       // 32768 (-> 95600)
    int*   sel_raw    = (int*)(ws + 95600);       // 512
    int*   sel_sorted = (int*)(ws + 96112);       // 512

    // 1. width prior
    k_width<<<MAXW, 256, 0, stream>>>(wemb, Ww1, bw1, ww2, bw2, wscore);
    // 2. init scores with b2 + width prior
    k_init_scores<<<(MCAND + 255) / 256, 256, 0, stream>>>(wscore, b2, widx, scores);
    // 3. fused span-score GEMM (bf16 WMMA)
    dim3 ggrid((MCAND + MBLK - 1) / MBLK, (NDIM + NBLK - 1) / NBLK);  // 235 x 8
    k_span_gemm<<<ggrid, 256, 0, stream>>>(emb, W1, b1, w2, scores);
    // 4. argsort descending by score (bitonic over 32768)
    k_sort_init<<<SORT_N / 256, 256, 0, stream>>>(scores, keys, vals);
    for (unsigned k = 2; k <= SORT_N; k <<= 1)
        for (unsigned j = k >> 1; j > 0; j >>= 1)
            k_bitonic<<<SORT_N / 256, 256, 0, stream>>>(keys, vals, j, k);
    // 5. greedy non-crossing selection (single wave32)
    k_greedy<<<1, 32, 0, stream>>>(vals, starts, ends, sel_raw);
    // 6. stable sort selected spans by (start, end)
    k_ranksort<<<1, 512, 0, stream>>>(sel_raw, starts, ends, sel_sorted);
    // 7. gather outputs
    k_gather<<<NTOP, 256, 0, stream>>>(sel_sorted, emb, scores, starts, ends, spk,
                                       (float*)d_out);
}
